// GraphTransformerLayer_90288802496850
// MI455X (gfx1250) — compile-verified, matrix operand order checked by reading, not run on hardware
//
#include <hip/hip_runtime.h>
#include <cmath>

// ---------------------------------------------------------------------------
// GraphTransformerLayer forward for MI455X (gfx1250, wave32, WMMA)
// B=16, N=512, D=512, H=8, HD=64, NBT=5, FF=2048
// ---------------------------------------------------------------------------

#define TOK   8192      // B*N
#define DIM   512
#define FF    2048
#define NHEAD 8
#define SEQ   512
#define BATCH 16

typedef __attribute__((ext_vector_type(16))) _Float16 v16h;
typedef __attribute__((ext_vector_type(8)))  _Float16 v8h;
typedef __attribute__((ext_vector_type(8)))  float    v8f;

union Frag16 { v16h v; v8h c[2]; _Float16 h[16]; };
union Acc8   { v8f  v; float f[8]; };

// Async global->LDS DMA path (gfx1250, ASYNCcnt-tracked), guarded so the
// source still compiles on toolchains without the builtins.
#if __has_builtin(__builtin_amdgcn_global_load_async_to_lds_b128) && \
    __has_builtin(__builtin_amdgcn_s_wait_asynccnt)
#define USE_ASYNC_LDS 1
typedef int v4i_ __attribute__((vector_size(16)));           // b128 payload type
typedef __attribute__((address_space(1))) v4i_ gv4i;         // global
typedef __attribute__((address_space(3))) v4i_ lv4i;         // LDS
__device__ __forceinline__ void async_ld_b128(const _Float16* g, _Float16* l) {
  __builtin_amdgcn_global_load_async_to_lds_b128(
      (gv4i*)(_Float16*)g, (lv4i*)l, 0, 0);
}
#else
#define USE_ASYNC_LDS 0
#endif

// ISA 7.12.2, 16-bit A/B fragments: lane -> M (A) / N (B), element e -> K with
//   K = kbase + (e&7) + 16*(e>>3),  kbase = 8*(lane>=16)
// => elements 0..7 and 8..15 are two contiguous 16-byte runs at kbase, kbase+16.
__device__ __forceinline__ v16h frag_ld(const _Float16* rowp, int kbase) {
  Frag16 f;
  f.c[0] = *(const v8h*)(rowp + kbase);
  f.c[1] = *(const v8h*)(rowp + kbase + 16);
  return f.v;
}

// ---------------------------------------------------------------------------
// LayerNorm (one block per token row of 512), writes f16
// ---------------------------------------------------------------------------
__global__ __launch_bounds__(256)
void ln_kernel(const float* __restrict__ x, const float* __restrict__ g,
               const float* __restrict__ beta, _Float16* __restrict__ out) {
  __shared__ float red[256];
  const int row = blockIdx.x;
  const int tid = threadIdx.x;
  const float* xr = x + (size_t)row * DIM;
  float a0 = xr[tid], a1 = xr[tid + 256];
  red[tid] = a0 + a1;
  __syncthreads();
  for (int o = 128; o > 0; o >>= 1) { if (tid < o) red[tid] += red[tid + o]; __syncthreads(); }
  const float mu = red[0] * (1.0f / DIM);
  __syncthreads();
  const float d0 = a0 - mu, d1 = a1 - mu;
  red[tid] = d0 * d0 + d1 * d1;
  __syncthreads();
  for (int o = 128; o > 0; o >>= 1) { if (tid < o) red[tid] += red[tid + o]; __syncthreads(); }
  const float rs = rsqrtf(red[0] * (1.0f / DIM) + 1e-5f);
  _Float16* orow = out + (size_t)row * DIM;
  orow[tid]       = (_Float16)(d0 * rs * g[tid]       + beta[tid]);
  orow[tid + 256] = (_Float16)(d1 * rs * g[tid + 256] + beta[tid + 256]);
}

// ---------------------------------------------------------------------------
// WMMA GEMM:  C[M,NC] = A(f16, MxK) * W(f32->f16, KxNC) + bias, epilogue.
// 256 thr = 8 waves; block tile 128x128; wave tile 32x64 (2x4 subtiles,
// 8 wmma / K-step from 2 A-frags + 4 B-frags). A staged via async DMA to LDS
// (when available); W staged transposed so all fragment reads are b128 loads.
// EPI 0: f16   1: exact GELU -> f16   2: +residual -> f32   3: (+res)*rowmask -> f32
// ---------------------------------------------------------------------------
template <int EPI>
__global__ __launch_bounds__(256)
void gemm_wmma(const _Float16* __restrict__ A, const float* __restrict__ W,
               const float* __restrict__ bias, const float* __restrict__ residual,
               const float* __restrict__ rowmask, float* __restrict__ outF,
               _Float16* __restrict__ outH, int M, int K, int NC) {
  __shared__ __align__(16) _Float16 As[128][40];    // [row][k]   (10 KB)
  __shared__ __align__(16) _Float16 Wst[128][40];   // [col][k]   (10 KB, transposed)

  const int tid  = threadIdx.x;
  const int lane = tid & 31;
  const int wv   = tid >> 5;
  const int rowBlk = blockIdx.y * 128;
  const int colBlk = blockIdx.x * 128;
  const int wr = wv >> 1;                 // row group 0..3  (32 rows)
  const int wc = wv & 1;                  // col group 0..1  (64 cols)
  const int r  = lane & 15;               // A row / B col within subtile
  const int kbase = (lane & 16) >> 1;

  Acc8 acc[2][4];
  #pragma unroll
  for (int t = 0; t < 2; t++)
    #pragma unroll
    for (int c = 0; c < 4; c++) acc[t][c].v = (v8f){};

  // staging coordinates
  const int ai = tid >> 1, aj = (tid & 1) * 16;   // A: 128 rows x 32 k
  const int wi = tid >> 3, wj = (tid & 7) * 16;   // W: 32 k x 128 cols

  for (int k0 = 0; k0 < K; k0 += 32) {
    const _Float16* ga = A + (size_t)(rowBlk + ai) * K + k0 + aj;
#if USE_ASYNC_LDS
    async_ld_b128(ga,     &As[ai][aj]);           // DMA straight into LDS
    async_ld_b128(ga + 8, &As[ai][aj + 8]);
#else
    *(v8h*)&As[ai][aj]     = *(const v8h*)(ga);
    *(v8h*)&As[ai][aj + 8] = *(const v8h*)(ga + 8);
#endif
    const float* gw = W + (size_t)(k0 + wi) * NC + colBlk + wj;
    #pragma unroll
    for (int q = 0; q < 16; q++) Wst[wj + q][wi] = (_Float16)gw[q];
    if (k0 + 32 < K) {                       // overlap next tile with WMMA pipe
      __builtin_prefetch(ga + 32, 0, 1);
      __builtin_prefetch(W + (size_t)(k0 + 32 + wi) * NC + colBlk + wj, 0, 1);
    }
#if USE_ASYNC_LDS
    __builtin_amdgcn_s_wait_asynccnt(0);     // LDS visible before barrier
#endif
    __syncthreads();

    Frag16 af[2], bf[4];
    #pragma unroll
    for (int t = 0; t < 2; t++) af[t].v = frag_ld(&As[wr * 32 + t * 16 + r][0], kbase);
    #pragma unroll
    for (int c = 0; c < 4; c++) bf[c].v = frag_ld(&Wst[wc * 64 + c * 16 + r][0], kbase);
    #pragma unroll
    for (int t = 0; t < 2; t++)
      #pragma unroll
      for (int c = 0; c < 4; c++)
        acc[t][c].v = __builtin_amdgcn_wmma_f32_16x16x32_f16(
            false, af[t].v, false, bf[c].v, (short)0, acc[t][c].v, false, false);
    __syncthreads();
  }

  // epilogue (C layout: lane -> N = lane&15, VGPR r8 -> M = r8 + 8*(lane>=16))
  const int n   = lane & 15;
  const int mh0 = (lane & 16) >> 1;
  #pragma unroll
  for (int t = 0; t < 2; t++) {
    #pragma unroll
    for (int c = 0; c < 4; c++) {
      const int col = colBlk + wc * 64 + c * 16 + n;
      const float bcol = bias[col];
      #pragma unroll
      for (int r8 = 0; r8 < 8; r8++) {
        const int row = rowBlk + wr * 32 + t * 16 + r8 + mh0;
        const size_t idx = (size_t)row * NC + col;
        float cval = acc[t][c].f[r8] + bcol;
        if (EPI == 0) {
          outH[idx] = (_Float16)cval;
        } else if (EPI == 1) {
          float gl = 0.5f * cval * (1.0f + erff(cval * 0.70710678118654752f));
          outH[idx] = (_Float16)gl;
        } else if (EPI == 2) {
          outF[idx] = residual[idx] + cval;
        } else {
          outF[idx] = (residual[idx] + cval) * rowmask[row];
        }
      }
    }
  }
}

// ---------------------------------------------------------------------------
// Attention: one block per (b, h, 16-query tile); 128 threads = 4 waves.
// Phase 1: S = Q K^T (WMMA, waves split 32 key tiles; frags loaded as b128)
// Phase 2: masked softmax with edge_bias[adj] gather (wave32 shfl reductions)
// Phase 3: O = P V with V tile staged transposed in LDS (contiguous frags)
// ---------------------------------------------------------------------------
__global__ __launch_bounds__(128)
void attn_kernel(const _Float16* __restrict__ Qm, const _Float16* __restrict__ Km,
                 const _Float16* __restrict__ Vm, const int* __restrict__ adj,
                 const float* __restrict__ mask, const float* __restrict__ ebias,
                 _Float16* __restrict__ O) {
  __shared__ __align__(16) float    S[16][520];    // raw scores     (33.3 KB)
  __shared__ __align__(16) _Float16 P[16][520];    // probabilities  (16.6 KB)
  __shared__ __align__(16) _Float16 Vst[64][40];   // V^T tile       ( 5.1 KB)

  const int qt = blockIdx.x, h = blockIdx.y, b = blockIdx.z;
  const int tid = threadIdx.x, lane = tid & 31, wv = tid >> 5;
  const int qr  = lane & 15;
  const int kbase = (lane & 16) >> 1;
  const int mh0 = kbase;
  const size_t rowQ0 = (size_t)b * SEQ + qt * 16;

  // Q fragments for this 16-query tile (HD=64 -> two k=32 chunks), reused.
  const _Float16* Qb = Qm + rowQ0 * DIM + h * 64;
  Frag16 q0, q1;
  q0.v = frag_ld(Qb + (size_t)qr * DIM, kbase);
  q1.v = frag_ld(Qb + (size_t)qr * DIM + 32, kbase);

  // Phase 1: scores
  for (int kt = wv; kt < 32; kt += 4) {
    const _Float16* Kb = Km + ((size_t)b * SEQ + kt * 16) * DIM + h * 64;
    Frag16 k0, k1;
    k0.v = frag_ld(Kb + (size_t)qr * DIM, kbase);       // lane -> key column
    k1.v = frag_ld(Kb + (size_t)qr * DIM + 32, kbase);
    Acc8 acc; acc.v = (v8f){};
    acc.v = __builtin_amdgcn_wmma_f32_16x16x32_f16(false, q0.v, false, k0.v,
                                                   (short)0, acc.v, false, false);
    acc.v = __builtin_amdgcn_wmma_f32_16x16x32_f16(false, q1.v, false, k1.v,
                                                   (short)0, acc.v, false, false);
    #pragma unroll
    for (int r = 0; r < 8; r++) S[r + mh0][kt * 16 + qr] = acc.f[r];
  }
  __syncthreads();

  // Phase 2: bias + mask + softmax (wave per row, 16 elems/lane)
  const float scale = 0.125f;                     // 1/sqrt(64)
  for (int row = wv; row < 16; row += 4) {
    const size_t q = rowQ0 + row;
    const float mq = mask[q];
    const int* adjRow = adj + q * SEQ;            // (b*N + qi)*N
    float vals[16];
    float vmax = -__builtin_inff();
    #pragma unroll
    for (int i = 0; i < 16; i++) {
      const int key = lane + i * 32;
      const float mk = mask[(size_t)b * SEQ + key];
      float v;
      if (mq == 0.0f || mk == 0.0f) v = -__builtin_inff();
      else v = S[row][key] * scale + ebias[adjRow[key] * NHEAD + h];
      vals[i] = v;
      vmax = fmaxf(vmax, v);
    }
    #pragma unroll
    for (int off = 16; off > 0; off >>= 1) vmax = fmaxf(vmax, __shfl_xor(vmax, off, 32));
    float sum = 0.0f;
    #pragma unroll
    for (int i = 0; i < 16; i++) {
      float p = (vmax == -__builtin_inff()) ? 0.0f : __expf(vals[i] - vmax);
      vals[i] = p; sum += p;
    }
    #pragma unroll
    for (int off = 16; off > 0; off >>= 1) sum += __shfl_xor(sum, off, 32);
    const float inv = (sum > 0.0f) ? 1.0f / sum : 0.0f;
    #pragma unroll
    for (int i = 0; i < 16; i++) P[row][lane + i * 32] = (_Float16)(vals[i] * inv);
  }
  __syncthreads();

  // Phase 3: O = P @ V ; wave wv owns hd columns [wv*16, wv*16+16).
  // Stage each 32-key x 64-hd V tile transposed so fragments are b128 loads.
  Acc8 acc; acc.v = (v8f){};
  const int n  = lane & 15;
  const int vi = tid >> 2;              // key row   0..31
  const int vj = (tid & 3) * 16;        // hd chunk  0/16/32/48
  for (int it = 0; it < 16; it++) {     // K = 512 in k=32 steps
    const _Float16* gv = Vm + ((size_t)b * SEQ + it * 32 + vi) * DIM + h * 64 + vj;
    #pragma unroll
    for (int q = 0; q < 16; q++) Vst[vj + q][vi] = gv[q];
    __syncthreads();
    Frag16 a, bb;
    a.v  = frag_ld(&P[qr][it * 32], kbase);
    bb.v = frag_ld(&Vst[wv * 16 + n][0], kbase);
    acc.v = __builtin_amdgcn_wmma_f32_16x16x32_f16(false, a.v, false, bb.v,
                                                   (short)0, acc.v, false, false);
    __syncthreads();
  }
  #pragma unroll
  for (int r = 0; r < 8; r++) {
    const size_t row = rowQ0 + r + mh0;
    O[row * DIM + h * 64 + wv * 16 + n] = (_Float16)acc.f[r];
  }
}

// ---------------------------------------------------------------------------
// Host launcher
// ---------------------------------------------------------------------------
extern "C" void kernel_launch(void* const* d_in, const int* in_sizes, int n_in,
                              void* d_out, int out_size, void* d_ws, size_t ws_size,
                              hipStream_t stream) {
  const float* x     = (const float*)d_in[0];
  const int*   adj   = (const int*)  d_in[1];
  const float* mask  = (const float*)d_in[2];
  const float* Wq    = (const float*)d_in[3];
  const float* bq    = (const float*)d_in[4];
  const float* Wk    = (const float*)d_in[5];
  const float* bk    = (const float*)d_in[6];
  const float* Wv    = (const float*)d_in[7];
  const float* bv    = (const float*)d_in[8];
  const float* Wo    = (const float*)d_in[9];
  const float* bo    = (const float*)d_in[10];
  const float* ebias = (const float*)d_in[11];
  const float* W1    = (const float*)d_in[12];
  const float* b1    = (const float*)d_in[13];
  const float* W2    = (const float*)d_in[14];
  const float* b2    = (const float*)d_in[15];
  const float* g1    = (const float*)d_in[16];
  const float* be1   = (const float*)d_in[17];
  const float* g2    = (const float*)d_in[18];
  const float* be2   = (const float*)d_in[19];
  float* out = (float*)d_out;

  // workspace carve-up (all offsets multiples of big powers of two)
  char* ws = (char*)d_ws;
  const size_t szH = (size_t)TOK * DIM * sizeof(_Float16);   // 8 MB
  _Float16* h1 = (_Float16*)ws;              ws += szH;
  _Float16* Qm = (_Float16*)ws;              ws += szH;
  _Float16* Km = (_Float16*)ws;              ws += szH;
  _Float16* Vm = (_Float16*)ws;              ws += szH;
  _Float16* Oa = (_Float16*)ws;              ws += szH;
  float*    x2 = (float*)ws;                 ws += (size_t)TOK * DIM * sizeof(float);
  _Float16* h2 = (_Float16*)ws;              ws += szH;
  _Float16* U  = (_Float16*)ws;              // TOK x FF f16 (32 MB)

  const dim3 blk256(256);
  const dim3 gridD(DIM / 128, TOK / 128);    // (4, 64)
  const dim3 gridF(FF / 128, TOK / 128);     // (16, 64)

  // 1. LN1 -> h1 (f16)
  ln_kernel<<<TOK, blk256, 0, stream>>>(x, g1, be1, h1);
  // 2-4. Q, K, V projections
  gemm_wmma<0><<<gridD, blk256, 0, stream>>>(h1, Wq, bq, nullptr, nullptr, nullptr, Qm, TOK, DIM, DIM);
  gemm_wmma<0><<<gridD, blk256, 0, stream>>>(h1, Wk, bk, nullptr, nullptr, nullptr, Km, TOK, DIM, DIM);
  gemm_wmma<0><<<gridD, blk256, 0, stream>>>(h1, Wv, bv, nullptr, nullptr, nullptr, Vm, TOK, DIM, DIM);
  // 5. attention
  attn_kernel<<<dim3(SEQ / 16, NHEAD, BATCH), dim3(128), 0, stream>>>(
      Qm, Km, Vm, adj, mask, ebias, Oa);
  // 6. output projection + residual -> x2 (f32)
  gemm_wmma<2><<<gridD, blk256, 0, stream>>>(Oa, Wo, bo, x, nullptr, x2, nullptr, TOK, DIM, DIM);
  // 7. LN2 -> h2
  ln_kernel<<<TOK, blk256, 0, stream>>>(x2, g2, be2, h2);
  // 8. FFN up + exact GELU -> U (f16)
  gemm_wmma<1><<<gridF, blk256, 0, stream>>>(h2, W1, b1, nullptr, nullptr, nullptr, U, TOK, DIM, FF);
  // 9. FFN down + residual + row mask -> out (f32)
  gemm_wmma<3><<<gridD, blk256, 0, stream>>>(U, W2, b2, x2, mask, out, nullptr, TOK, FF, DIM);
}
